// EdgeAwareGATEncoder_80745385165159
// MI455X (gfx1250) — matrix operand order repaired and checked by reading, not compiled, not run on hardware
//
#include <hip/hip_runtime.h>
#include <hip/hip_bf16.h>
#include <string.h>

// ---------------------------------------------------------------------------
// EdgeAwareGATEncoder for MI455X (gfx1250), wave32, WMMA bf16 16x16x32.
// B=4, L=2048, K=30, HID=128, EDGE=16, NL=3, NH=4, HD=32.
// ---------------------------------------------------------------------------

#define B_   4
#define L_   2048
#define K_   30
#define HID_ 128
#define EDGE_ 16
#define NL_  3
#define NH_  4
#define HD_  32
#define M_   (B_ * L_)          // 8192 rows
#define SCALE_ 0.17677669529663687f   // 32^-0.5
#define LN_EPS_ 1e-5f

typedef __attribute__((ext_vector_type(16))) __bf16 v16bf;
typedef __attribute__((ext_vector_type(8)))  __bf16 v8bf;
typedef __attribute__((ext_vector_type(8)))  float  v8f;

// fp32 -> bf16 round-to-nearest-even
static __device__ __forceinline__ __bf16 f2bf(float f) {
    unsigned u = __builtin_bit_cast(unsigned, f);
    unsigned r = u + 0x7FFFu + ((u >> 16) & 1u);
    unsigned short h = (unsigned short)(r >> 16);
    return __builtin_bit_cast(__bf16, h);
}

// ---------------------------------------------------------------------------
// prep: bf16-convert + transpose weights (Wt[n][k] = W[k][n]) and pack biases.
//   Wt_qkv : [NL][3][128][128] bf16   (q,k,v)
//   Wt_o   : [NL][128][128]   bf16
//   b_qkv  : [NL][3][128]     f32
// ---------------------------------------------------------------------------
__global__ __launch_bounds__(256) void prep_kernel(
    const float* __restrict__ Wq, const float* __restrict__ Wk,
    const float* __restrict__ Wv, const float* __restrict__ Wo,
    const float* __restrict__ bq, const float* __restrict__ bk,
    const float* __restrict__ bv,
    __bf16* __restrict__ Wt_qkv, __bf16* __restrict__ Wt_o,
    float* __restrict__ b_qkv)
{
    int tid = blockIdx.x * blockDim.x + threadIdx.x;
    const int WELEM = HID_ * HID_;                 // 16384
    const int TOTAL = NL_ * 4 * WELEM;             // q,k,v,o
    if (tid < TOTAL) {
        int l = tid / (4 * WELEM);
        int r = tid % (4 * WELEM);
        int m = r / WELEM;
        int e = r % WELEM;
        int n = e / HID_;          // output column (dest-major)
        int k = e % HID_;          // reduction index
        const float* W = (m == 0) ? Wq : (m == 1) ? Wk : (m == 2) ? Wv : Wo;
        __bf16 v = f2bf(W[l * WELEM + k * HID_ + n]);
        if (m < 3) Wt_qkv[(size_t)(l * 3 + m) * WELEM + n * HID_ + k] = v;
        else       Wt_o[(size_t)l * WELEM + n * HID_ + k] = v;
    }
    if (tid < NL_ * 3 * HID_) {
        int l = tid / (3 * HID_);
        int r = tid % (3 * HID_);
        int m = r / HID_;
        int c = r % HID_;
        const float* bb = (m == 0) ? bq : (m == 1) ? bk : bv;
        b_qkv[tid] = bb[l * HID_ + c];
    }
}

// ---------------------------------------------------------------------------
// gemm_bf16: out[y] = A @ W[y]^T(stored [N][K]) + bias[y]
//   A    : [M][128] f32 (converted to bf16 on the fly)
//   Wt   : base of bf16 [gridDim.y][128][128] dest-major
//   bias : base of f32  [gridDim.y][128]
//   out  : base of f32  [gridDim.y][M][128]
// Block: 128 threads (4 waves). Tile: 32 rows x 128 cols.
// Wave w: rows 0..31, cols [w*32, w*32+32). 16 WMMAs / wave.
// All 16 fragments are loaded into distinct registers first so the WMMAs can
// issue back-to-back with partial s_wait_dscnt instead of full-drain waits.
// ---------------------------------------------------------------------------
__global__ __launch_bounds__(128) void gemm_bf16_kernel(
    const float* __restrict__ A,
    const __bf16* __restrict__ Wt,
    const float* __restrict__ bias,
    float* __restrict__ out, int M)
{
    __shared__ __align__(64) __bf16 Wsh[HID_ * HID_];   // 32 KB
    __shared__ __align__(64) __bf16 Ash[32 * HID_];     //  8 KB

    const __bf16* Wp = Wt  + (size_t)blockIdx.y * HID_ * HID_;
    const float*  bp = bias + (size_t)blockIdx.y * HID_;
    float*        op = out + (size_t)blockIdx.y * (size_t)M * HID_;

    const int t  = threadIdx.x;
    const int mb = blockIdx.x * 32;

    // stage full weight matrix (2048 uint4, 16 per thread)
    {
        const uint4* ws = (const uint4*)Wp;
        uint4* wd = (uint4*)Wsh;
        #pragma unroll
        for (int j = 0; j < 16; ++j) wd[j * 128 + t] = ws[j * 128 + t];
    }
    // stage + convert A tile: 32x128 f32 = 1024 float4, 8 per thread
    {
        const float4* as = (const float4*)(A + (size_t)mb * HID_);
        #pragma unroll
        for (int j = 0; j < 8; ++j) {
            int e4 = j * 128 + t;            // float4 index within tile
            float4 f = as[e4];
            __bf16* d = Ash + (e4 << 2);
            d[0] = f2bf(f.x); d[1] = f2bf(f.y);
            d[2] = f2bf(f.z); d[3] = f2bf(f.w);
        }
    }
    __syncthreads();

    const int w    = t >> 5;
    const int lane = t & 31;
    const int lm   = lane & 15;
    const int hi   = lane >> 4;
    const int nb   = w * 32;

    // ---- load ALL fragments first (16 frags x 8 VGPR = 128 VGPRs) ----
    v16bf afrag[2][4];   // [row-tile][k-step]
    v16bf bfrag[2][4];   // [col-tile][k-step]
    #pragma unroll
    for (int ks = 0; ks < 4; ++ks) {
        const int k0 = ks * 32;
        // A fragments: lane = M, contiguous K = k0 + hi*16 + {0..15}
        v8bf a0lo = *(const v8bf*)(Ash + (0  + lm) * HID_ + k0 + hi * 16);
        v8bf a0hi = *(const v8bf*)(Ash + (0  + lm) * HID_ + k0 + hi * 16 + 8);
        v8bf a1lo = *(const v8bf*)(Ash + (16 + lm) * HID_ + k0 + hi * 16);
        v8bf a1hi = *(const v8bf*)(Ash + (16 + lm) * HID_ + k0 + hi * 16 + 8);
        afrag[0][ks] = __builtin_shufflevector(a0lo, a0hi, 0,1,2,3,4,5,6,7,8,9,10,11,12,13,14,15);
        afrag[1][ks] = __builtin_shufflevector(a1lo, a1hi, 0,1,2,3,4,5,6,7,8,9,10,11,12,13,14,15);
        // B fragments: lane = N, contiguous K = k0 + hi*16 + {0..15}
        bfrag[0][ks] = *(const v16bf*)(Wsh + (nb + lm)      * HID_ + k0 + hi * 16);
        bfrag[1][ks] = *(const v16bf*)(Wsh + (nb + 16 + lm) * HID_ + k0 + hi * 16);
    }

    // ---- 16 back-to-back WMMAs ----
    v8f acc00 = {}, acc01 = {}, acc10 = {}, acc11 = {};
    #pragma unroll
    for (int ks = 0; ks < 4; ++ks) {
        acc00 = __builtin_amdgcn_wmma_f32_16x16x32_bf16(false, afrag[0][ks], false, bfrag[0][ks], (short)0, acc00, false, false);
        acc01 = __builtin_amdgcn_wmma_f32_16x16x32_bf16(false, afrag[0][ks], false, bfrag[1][ks], (short)0, acc01, false, false);
        acc10 = __builtin_amdgcn_wmma_f32_16x16x32_bf16(false, afrag[1][ks], false, bfrag[0][ks], (short)0, acc10, false, false);
        acc11 = __builtin_amdgcn_wmma_f32_16x16x32_bf16(false, afrag[1][ks], false, bfrag[1][ks], (short)0, acc11, false, false);
    }

    // C layout: VGPR v, lane l -> row = v + (l>>4)*8, col = l&15
    float bias0 = bp[nb + lm];
    float bias1 = bp[nb + 16 + lm];
    #pragma unroll
    for (int v = 0; v < 8; ++v) {
        int r0 = mb + v + hi * 8;
        int r1 = r0 + 16;
        int c0 = nb + lm, c1 = c0 + 16;
        op[(size_t)r0 * HID_ + c0] = acc00[v] + bias0;
        op[(size_t)r0 * HID_ + c1] = acc01[v] + bias1;
        op[(size_t)r1 * HID_ + c0] = acc10[v] + bias0;
        op[(size_t)r1 * HID_ + c1] = acc11[v] + bias1;
    }
}

// ---------------------------------------------------------------------------
// attention: one block per node (b,l); one wave per head; lane = HD channel.
//   qkv : [3][M][128] f32 planes (q, kf, vf) — kf/vf are per-node projections,
//         gathered here (equivalent to projecting after gather).
// ---------------------------------------------------------------------------
__global__ __launch_bounds__(128) void attn_kernel(
    const float* __restrict__ qkv,
    const float* __restrict__ h_edges,   // [B][L][K][EDGE]
    const int*   __restrict__ eidx,      // [B][L][K]
    const float* __restrict__ mask,      // [B][L]
    const float* __restrict__ We,        // [EDGE][NH] (layer slice)
    const float* __restrict__ be,        // [NH]
    float* __restrict__ o)               // [M][128]
{
    const int node = blockIdx.x;         // 0..M-1
    const int b    = node >> 11;         // L = 2048
    const int lane = threadIdx.x & 31;
    const int h    = threadIdx.x >> 5;   // head

    const float* qp = qkv + (size_t)node * HID_;
    const float* kf = qkv + (size_t)1 * M_ * HID_;
    const float* vf = qkv + (size_t)2 * M_ * HID_;

    __shared__ int   idx_sh[K_];
    __shared__ float w_sh[NH_][32];

    if (threadIdx.x < K_) idx_sh[threadIdx.x] = eidx[(size_t)node * K_ + threadIdx.x];
    __syncthreads();

    // warm L2/L0 for the gathered K/V rows: lane kk prefetches neighbor kk's
    // row segment for this head (global_prefetch_b8; 30 lines per matrix).
    if (lane < K_) {
        size_t roff = ((size_t)(b * L_ + idx_sh[lane])) * HID_ + h * HD_;
        __builtin_prefetch(kf + roff, 0, 3);
        __builtin_prefetch(vf + roff, 0, 3);
    }

    const float qd      = qp[h * HD_ + lane];
    const float mask_l  = mask[node];

    // per-lane (= per-neighbor) edge bias + neighbor mask
    float ebias = 0.f, nmask = 0.f;
    if (lane < K_) {
        const float* ep = h_edges + ((size_t)node * K_ + lane) * EDGE_;
        float s = be[h];
        #pragma unroll
        for (int e = 0; e < EDGE_; ++e) s += ep[e] * We[e * NH_ + h];
        ebias = s;
        nmask = mask[b * L_ + idx_sh[lane]];
    }

    // scores: q . k_j reduced across the wave; lane kk keeps neighbor kk
    float myscore = -1e30f;
    for (int kk = 0; kk < K_; ++kk) {
        int j = idx_sh[kk];
        float p = qd * kf[((size_t)(b * L_ + j)) * HID_ + h * HD_ + lane];
        p += __shfl_xor(p, 16, 32);
        p += __shfl_xor(p,  8, 32);
        p += __shfl_xor(p,  4, 32);
        p += __shfl_xor(p,  2, 32);
        p += __shfl_xor(p,  1, 32);
        if (lane == kk) myscore = p * SCALE_;
    }
    if (lane < K_) {
        myscore += ebias;
        if (mask_l * nmask == 0.f) myscore = -1e9f;
    }

    // stable softmax over lanes 0..K-1 (lanes >= K contribute 0)
    float m = myscore;
    m = fmaxf(m, __shfl_xor(m, 16, 32));
    m = fmaxf(m, __shfl_xor(m,  8, 32));
    m = fmaxf(m, __shfl_xor(m,  4, 32));
    m = fmaxf(m, __shfl_xor(m,  2, 32));
    m = fmaxf(m, __shfl_xor(m,  1, 32));
    float e = (lane < K_) ? __expf(myscore - m) : 0.f;
    float s = e;
    s += __shfl_xor(s, 16, 32);
    s += __shfl_xor(s,  8, 32);
    s += __shfl_xor(s,  4, 32);
    s += __shfl_xor(s,  2, 32);
    s += __shfl_xor(s,  1, 32);
    w_sh[h][lane] = e / s;
    __syncthreads();

    // weighted sum of gathered V rows
    float acc = 0.f;
    for (int kk = 0; kk < K_; ++kk) {
        int j = idx_sh[kk];
        acc += w_sh[h][kk] * vf[((size_t)(b * L_ + j)) * HID_ + h * HD_ + lane];
    }
    o[(size_t)node * HID_ + h * HD_ + lane] = acc;
}

// ---------------------------------------------------------------------------
// residual + LayerNorm + mask. One wave per row (lane holds 4 channels).
// ---------------------------------------------------------------------------
__global__ __launch_bounds__(128) void ln_kernel(
    const float* __restrict__ hin,   // [M][128]
    const float* __restrict__ y,     // [M][128] (o@Wo + bo)
    const float* __restrict__ g,     // [128]
    const float* __restrict__ bta,   // [128]
    const float* __restrict__ mask,  // [M]
    float* __restrict__ hout)        // [M][128]
{
    const int row  = blockIdx.x * 4 + (threadIdx.x >> 5);
    const int lane = threadIdx.x & 31;
    const int c    = lane * 4;

    float4 hv = *(const float4*)(hin + (size_t)row * HID_ + c);
    float4 yv = *(const float4*)(y   + (size_t)row * HID_ + c);
    float x0 = hv.x + yv.x, x1 = hv.y + yv.y, x2 = hv.z + yv.z, x3 = hv.w + yv.w;

    float s  = x0 + x1 + x2 + x3;
    float s2 = x0 * x0 + x1 * x1 + x2 * x2 + x3 * x3;
    #pragma unroll
    for (int d = 16; d > 0; d >>= 1) {
        s  += __shfl_xor(s,  d, 32);
        s2 += __shfl_xor(s2, d, 32);
    }
    const float inv_n = 1.0f / (float)HID_;
    float mu  = s * inv_n;
    float var = s2 * inv_n - mu * mu;
    float inv = rsqrtf(var + LN_EPS_);
    float mk  = mask[row];

    float4 gv = *(const float4*)(g   + c);
    float4 bv = *(const float4*)(bta + c);
    float4 r;
    r.x = ((x0 - mu) * inv * gv.x + bv.x) * mk;
    r.y = ((x1 - mu) * inv * gv.y + bv.y) * mk;
    r.z = ((x2 - mu) * inv * gv.z + bv.z) * mk;
    r.w = ((x3 - mu) * inv * gv.w + bv.w) * mk;
    *(float4*)(hout + (size_t)row * HID_ + c) = r;
}

// ---------------------------------------------------------------------------
// launch
// ---------------------------------------------------------------------------
extern "C" void kernel_launch(void* const* d_in, const int* in_sizes, int n_in,
                              void* d_out, int out_size, void* d_ws, size_t ws_size,
                              hipStream_t stream) {
    const float* h_nodes = (const float*)d_in[0];
    const float* h_edges = (const float*)d_in[1];
    const float* mask    = (const float*)d_in[2];
    const float* Wq      = (const float*)d_in[3];
    const float* bq      = (const float*)d_in[4];
    const float* Wk      = (const float*)d_in[5];
    const float* bk      = (const float*)d_in[6];
    const float* Wv      = (const float*)d_in[7];
    const float* bv      = (const float*)d_in[8];
    const float* We      = (const float*)d_in[9];
    const float* be      = (const float*)d_in[10];
    const float* Wo      = (const float*)d_in[11];
    const float* bo      = (const float*)d_in[12];
    const float* ln_g    = (const float*)d_in[13];
    const float* ln_b    = (const float*)d_in[14];
    const int*   eidx    = (const int*)d_in[15];

    // workspace carve-up (all 256B aligned); total ~30 MB
    size_t off = 0;
    char* base = (char*)d_ws;
    auto take = [&](size_t bytes) -> void* {
        void* p = base + off;
        off += (bytes + 255) & ~(size_t)255;
        return p;
    };
    __bf16* Wt_qkv = (__bf16*)take((size_t)NL_ * 3 * HID_ * HID_ * sizeof(__bf16));
    __bf16* Wt_o   = (__bf16*)take((size_t)NL_ * HID_ * HID_ * sizeof(__bf16));
    float*  b_qkv  = (float*) take((size_t)NL_ * 3 * HID_ * sizeof(float));
    float*  qkv    = (float*) take((size_t)3 * M_ * HID_ * sizeof(float));
    float*  obuf   = (float*) take((size_t)M_ * HID_ * sizeof(float));
    float*  ybuf   = (float*) take((size_t)M_ * HID_ * sizeof(float));
    float*  hA     = (float*) take((size_t)M_ * HID_ * sizeof(float));
    float*  hB     = (float*) take((size_t)M_ * HID_ * sizeof(float));

    // 1) one-time weight conversion (bf16, dest-major) + bias packing
    {
        int total = NL_ * 4 * HID_ * HID_;
        int blocks = (total + 255) / 256;
        prep_kernel<<<blocks, 256, 0, stream>>>(Wq, Wk, Wv, Wo, bq, bk, bv,
                                                Wt_qkv, Wt_o, b_qkv);
    }

    const float* hin = h_nodes;
    for (int i = 0; i < NL_; ++i) {
        // 2) fused QKV projections (WMMA bf16)
        gemm_bf16_kernel<<<dim3(M_ / 32, 3), 128, 0, stream>>>(
            hin,
            Wt_qkv + (size_t)i * 3 * HID_ * HID_,
            b_qkv + (size_t)i * 3 * HID_,
            qkv, M_);

        // 3) gather-attention with edge bias + masked softmax
        attn_kernel<<<M_, 128, 0, stream>>>(
            qkv, h_edges, eidx, mask,
            We + (size_t)i * EDGE_ * NH_,
            be + (size_t)i * NH_,
            obuf);

        // 4) output projection (WMMA bf16)
        gemm_bf16_kernel<<<dim3(M_ / 32, 1), 128, 0, stream>>>(
            obuf,
            Wt_o + (size_t)i * HID_ * HID_,
            bo + (size_t)i * HID_,
            ybuf, M_);

        // 5) residual + LayerNorm + mask
        float* hout = (i == NL_ - 1) ? (float*)d_out : ((i == 0) ? hA : hB);
        ln_kernel<<<M_ / 4, 128, 0, stream>>>(
            hin, ybuf,
            ln_g + (size_t)i * HID_,
            ln_b + (size_t)i * HID_,
            mask, hout);
        hin = hout;
    }
}
